// Agent_81758997446926
// MI455X (gfx1250) — matrix operand order repaired and checked by reading, not compiled, not run on hardware
//
#include <hip/hip_runtime.h>

typedef _Float16 h16v __attribute__((ext_vector_type(16)));
typedef _Float16 h8v  __attribute__((ext_vector_type(8)));
typedef float    f8v  __attribute__((ext_vector_type(8)));

#define B_TOTAL 131072
#define OBS     376
#define K1P     384      // OBS padded to 12 k-tiles of 32
#define H       64
#define AD      17
#define NPAIR   153      // 17*18/2 lower-triangle pairs
#define NPPAD   160      // padded to 10 n-tiles of 16
#define HB_STR  65       // hbuf row stride (odd -> conflict free)
#define GR_STR  161      // gram row stride (odd)
#define LOG2PI  1.8378770664093453f
#define EPS_RMS 1.19209290e-07f

// ---- workspace layout (in _Float16 elements) ----
#define WS_CW1  0
#define WS_AW1  (WS_CW1 + 64*K1P)
#define WS_NW1  (WS_AW1 + 64*K1P)
#define WS_CW2  (WS_NW1 + 64*K1P)
#define WS_AW2  (WS_CW2 + 64*64)
#define WS_NW2  (WS_AW2 + 64*64)
#define WS_NFW  (WS_NW2 + 64*64)
#define WS_P    (WS_NFW + 64*64)

// ---- LDS layout (bytes) ----
#define SM_XT    0                        // f16 [128][384] = 98304  (aliased by gram f32 [128][161])
#define SM_HB    98304                    // f32 [128][65]  = 33280
#define SM_MEAN  (SM_HB + 33280)          // f32 [128][17]  = 8704
#define SM_SIG2  (SM_MEAN + 8704)         // f16 [128][64]  = 16384
#define SM_VAL   (SM_SIG2 + 16384)        // f32 [128]      = 512
#define SM_BYTES (SM_VAL + 512)

__device__ __forceinline__ f8v wmma_f16(h16v a, h16v b, f8v c) {
    return __builtin_amdgcn_wmma_f32_16x16x32_f16(false, a, false, b, (short)0, c,
                                                  false, false);
}

// A from f16 LDS (row base pointer already includes the lane's M row).
template <int NKT, int NNT>
__device__ __forceinline__ void gemm_f16A(const _Float16* Abase, const _Float16* W,
                                          int wstride, int lane, f8v* acc) {
    const int kg = lane >> 4;
    const int nloc = lane & 15;
    #pragma unroll
    for (int kt = 0; kt < NKT; ++kt) {
        union { h16v v; h8v h2[2]; } u;
        const _Float16* p = Abase + kt * 32 + kg * 16;
        u.h2[0] = *(const h8v*)(p);
        u.h2[1] = *(const h8v*)(p + 8);
        #pragma unroll
        for (int nt = 0; nt < NNT; ++nt) {
            h16v b = *(const h16v*)(W + (nt * 16 + nloc) * wstride + kt * 32 + kg * 16);
            acc[nt] = wmma_f16(u.v, b, acc[nt]);
        }
    }
}

// A from f32 LDS (convert to f16 on the fly), K = 64 (2 k-tiles), N = 64.
__device__ __forceinline__ void gemm_f32A(const float* Abase, const _Float16* W,
                                          int lane, f8v* acc) {
    const int kg = lane >> 4;
    const int nloc = lane & 15;
    #pragma unroll
    for (int kt = 0; kt < 2; ++kt) {
        union { h16v v; _Float16 e[16]; } u;
        const float* p = Abase + kt * 32 + kg * 16;
        #pragma unroll
        for (int i = 0; i < 8; ++i) u.e[i] = (_Float16)p[i];
        #pragma unroll
        for (int i = 0; i < 8; ++i) u.e[8 + i] = (_Float16)p[8 + i];
        #pragma unroll
        for (int nt = 0; nt < 4; ++nt) {
            h16v b = *(const h16v*)(W + (nt * 16 + nloc) * 64 + kt * 32 + kg * 16);
            acc[nt] = wmma_f16(u.v, b, acc[nt]);
        }
    }
}

// D tile epilogue: bias + SiLU -> hbuf (f32, stride HB_STR).
__device__ __forceinline__ void store_silu(float* hb, int mrow0, int lane,
                                           const f8v* acc, const float* bias) {
    const int rh = (lane >> 4) * 8;
    const int c0 = lane & 15;
    #pragma unroll
    for (int nt = 0; nt < 4; ++nt)
        #pragma unroll
        for (int r = 0; r < 8; ++r) {
            int row = mrow0 + rh + r;
            int col = nt * 16 + c0;
            float v = acc[nt][r] + bias[col];
            v = v / (1.0f + __expf(-v));
            hb[row * HB_STR + col] = v;
        }
}

__device__ __forceinline__ void rmsnorm_row(float* hb, int r, const float* g) {
    float ss = 0.f;
    #pragma unroll
    for (int n = 0; n < H; ++n) { float v = hb[r * HB_STR + n]; ss += v * v; }
    float rs = rsqrtf(ss * (1.0f / H) + EPS_RMS);
    #pragma unroll
    for (int n = 0; n < H; ++n) hb[r * HB_STR + n] *= rs * g[n];
}

// ------------------ setup kernels ------------------
__global__ void prep_weights(const float* cW1, const float* aW1, const float* nW1,
                             const float* cW2, const float* aW2, const float* nW2,
                             const float* nfW, _Float16* ws) {
    int gid = blockIdx.x * blockDim.x + threadIdx.x;
    int stride = gridDim.x * blockDim.x;
    for (int i = gid; i < 3 * 64 * K1P; i += stride) {
        int w = i / (64 * K1P);
        int rem = i % (64 * K1P);
        int n = rem / K1P, k = rem % K1P;
        const float* src = (w == 0) ? cW1 : (w == 1) ? aW1 : nW1;
        float v = (k < OBS) ? src[n * OBS + k] : 0.f;
        ws[i] = (_Float16)v;
    }
    _Float16* w2 = ws + 3 * 64 * K1P;
    for (int i = gid; i < 4 * 64 * 64; i += stride) {
        int w = i / 4096;
        int rem = i % 4096;
        const float* src = (w == 0) ? cW2 : (w == 1) ? aW2 : (w == 2) ? nW2 : nfW;
        w2[i] = (_Float16)src[rem];
    }
}

__global__ void prep_P(const float* amW, _Float16* P) {
    __shared__ float nrm[AD];
    int tid = threadIdx.x;
    if (tid < AD) {
        float s = 0.f;
        for (int h = 0; h < H; ++h) { float v = amW[tid * H + h]; s += v * v; }
        nrm[tid] = sqrtf(s) + 1e-8f;
    }
    __syncthreads();
    for (int idx = tid; idx < NPPAD * H; idx += blockDim.x) {
        int p = idx / H, h = idx % H;
        float v = 0.f;
        if (p < NPAIR) {
            int i = 0;
            while ((i + 1) * (i + 2) / 2 <= p) ++i;
            int j = p - i * (i + 1) / 2;
            v = (amW[i * H + h] / nrm[i]) * (amW[j * H + h] / nrm[j]);
        }
        P[idx] = (_Float16)v;
    }
}

// ------------------ fused main kernel ------------------
__global__ __launch_bounds__(128) void fused_agent(
    const float* __restrict__ x, const float* __restrict__ action,
    const _Float16* cW1h, const float* cb1, const float* cg1,
    const _Float16* cW2h, const float* cb2, const float* cW3, const float* cb3,
    const _Float16* aW1h, const float* ab1,
    const _Float16* aW2h, const float* ab2,
    const float* amW, const float* amb,
    const _Float16* nW1h, const float* nb1, const float* ng1,
    const _Float16* nW2h, const float* nb2, const float* ng2,
    const _Float16* nfWh, const float* nfb,
    const _Float16* Ph, float* __restrict__ out) {
    extern __shared__ unsigned char smem[];
    _Float16* xT   = (_Float16*)(smem + SM_XT);
    float*    gram = (float*)(smem + SM_XT);   // aliases xT after trunks
    float*    hb   = (float*)(smem + SM_HB);
    float*    meanB= (float*)(smem + SM_MEAN);
    _Float16* sig2 = (_Float16*)(smem + SM_SIG2);
    float*    valB = (float*)(smem + SM_VAL);

    const int tid = threadIdx.x;
    const int lane = tid & 31;
    const int wave = tid >> 5;
    const long long rowBlock = (long long)blockIdx.x * 128;

    // ---- phase 0: stage x tile as f16 ----
    for (int idx = tid; idx < 128 * K1P; idx += 128) {
        int r = idx / K1P, c = idx % K1P;
        float v = (c < OBS) ? x[(rowBlock + r) * OBS + c] : 0.f;
        xT[idx] = (_Float16)v;
    }
    __syncthreads();

    const f8v z8 = {0, 0, 0, 0, 0, 0, 0, 0};

    // =================== CRITIC ===================
    #pragma unroll
    for (int mt = 0; mt < 2; ++mt) {
        int mrow0 = wave * 32 + mt * 16;
        f8v acc[4] = {z8, z8, z8, z8};
        gemm_f16A<12, 4>(xT + (mrow0 + (lane & 15)) * K1P, cW1h, K1P, lane, acc);
        store_silu(hb, mrow0, lane, acc, cb1);
    }
    __syncthreads();
    rmsnorm_row(hb, tid, cg1);
    __syncthreads();
    #pragma unroll
    for (int mt = 0; mt < 2; ++mt) {
        int mrow0 = wave * 32 + mt * 16;
        f8v acc[4] = {z8, z8, z8, z8};
        gemm_f32A(hb + (mrow0 + (lane & 15)) * HB_STR, cW2h, lane, acc);
        store_silu(hb, mrow0, lane, acc, cb2);
    }
    __syncthreads();
    {   // value head (thread-per-row dot)
        float v = cb3[0];
        #pragma unroll
        for (int n = 0; n < H; ++n) v += hb[tid * HB_STR + n] * cW3[n];
        valB[tid] = v;
    }
    __syncthreads();

    // =================== ACTOR ===================
    #pragma unroll
    for (int mt = 0; mt < 2; ++mt) {
        int mrow0 = wave * 32 + mt * 16;
        f8v acc[4] = {z8, z8, z8, z8};
        gemm_f16A<12, 4>(xT + (mrow0 + (lane & 15)) * K1P, aW1h, K1P, lane, acc);
        store_silu(hb, mrow0, lane, acc, ab1);
    }
    __syncthreads();
    #pragma unroll
    for (int mt = 0; mt < 2; ++mt) {
        int mrow0 = wave * 32 + mt * 16;
        f8v acc[4] = {z8, z8, z8, z8};
        gemm_f32A(hb + (mrow0 + (lane & 15)) * HB_STR, aW2h, lane, acc);
        store_silu(hb, mrow0, lane, acc, ab2);
    }
    __syncthreads();
    {   // action mean head (thread-per-row, 17x64)
        for (int a = 0; a < AD; ++a) {
            float m = amb[a];
            #pragma unroll
            for (int n = 0; n < H; ++n) m += hb[tid * HB_STR + n] * amW[a * H + n];
            meanB[tid * AD + a] = m;
        }
    }
    __syncthreads();

    // =================== NOISE ===================
    #pragma unroll
    for (int mt = 0; mt < 2; ++mt) {
        int mrow0 = wave * 32 + mt * 16;
        f8v acc[4] = {z8, z8, z8, z8};
        gemm_f16A<12, 4>(xT + (mrow0 + (lane & 15)) * K1P, nW1h, K1P, lane, acc);
        store_silu(hb, mrow0, lane, acc, nb1);
    }
    __syncthreads();
    rmsnorm_row(hb, tid, ng1);
    __syncthreads();
    #pragma unroll
    for (int mt = 0; mt < 2; ++mt) {
        int mrow0 = wave * 32 + mt * 16;
        f8v acc[4] = {z8, z8, z8, z8};
        gemm_f32A(hb + (mrow0 + (lane & 15)) * HB_STR, nW2h, lane, acc);
        store_silu(hb, mrow0, lane, acc, nb2);
    }
    __syncthreads();
    rmsnorm_row(hb, tid, ng2);
    __syncthreads();
    // noise fc -> log_sigma -> sigma^2 (f16) for the gram GEMM
    #pragma unroll
    for (int mt = 0; mt < 2; ++mt) {
        int mrow0 = wave * 32 + mt * 16;
        f8v acc[4] = {z8, z8, z8, z8};
        gemm_f32A(hb + (mrow0 + (lane & 15)) * HB_STR, nfWh, lane, acc);
        const int rh = (lane >> 4) * 8;
        const int c0 = lane & 15;
        #pragma unroll
        for (int nt = 0; nt < 4; ++nt)
            #pragma unroll
            for (int r = 0; r < 8; ++r) {
                int row = mrow0 + rh + r;
                int col = nt * 16 + c0;
                float ls = acc[nt][r] + nfb[col];
                float sg = __expf(fminf(ls, 5.0f));
                sig2[row * H + col] = (_Float16)(sg * sg);
            }
    }
    __syncthreads();   // xT no longer needed; gram buffer may alias it now

    // ========== GRAM as GEMM: gram[b,p] = sum_h sig2[b,h] * P[p,h] ==========
    #pragma unroll
    for (int mt = 0; mt < 2; ++mt) {
        int mrow0 = wave * 32 + mt * 16;
        f8v acc[10] = {z8, z8, z8, z8, z8, z8, z8, z8, z8, z8};
        gemm_f16A<2, 10>(sig2 + (mrow0 + (lane & 15)) * H, Ph, H, lane, acc);
        const int rh = (lane >> 4) * 8;
        const int c0 = lane & 15;
        #pragma unroll
        for (int nt = 0; nt < 10; ++nt)
            #pragma unroll
            for (int r = 0; r < 8; ++r) {
                int p = nt * 16 + c0;
                if (p < NPAIR) gram[(mrow0 + rh + r) * GR_STR + p] = acc[nt][r];
            }
    }
    __syncthreads();

    // ========== per-row Cholesky / solve / outputs ==========
    {
        const int r = tid;
        const long long gr = rowBlock + r;
        float* A = gram + r * GR_STR;   // packed lower: p(i,j) = i*(i+1)/2 + j
        #pragma unroll
        for (int j = 0; j < AD; ++j) A[j * (j + 1) / 2 + j] += 1e-4f;
        float logdiag = 0.f;
        #pragma unroll
        for (int j = 0; j < AD; ++j) {
            float d = A[j * (j + 1) / 2 + j];
            #pragma unroll
            for (int k = 0; k < j; ++k) { float l = A[j * (j + 1) / 2 + k]; d -= l * l; }
            float s = sqrtf(d);
            A[j * (j + 1) / 2 + j] = s;
            logdiag += __logf(s);
            float inv = 1.0f / s;
            #pragma unroll
            for (int i = j + 1; i < AD; ++i) {
                float t = A[i * (i + 1) / 2 + j];
                #pragma unroll
                for (int k = 0; k < j; ++k)
                    t -= A[i * (i + 1) / 2 + k] * A[j * (j + 1) / 2 + k];
                A[i * (i + 1) / 2 + j] = t * inv;
            }
        }
        float y[AD];
        float mahal = 0.f;
        #pragma unroll
        for (int i = 0; i < AD; ++i) {
            float a = action[gr * AD + i];
            out[gr * AD + i] = a;   // output 0: action passthrough
            float t = a - meanB[r * AD + i];
            #pragma unroll
            for (int k = 0; k < i; ++k) t -= A[i * (i + 1) / 2 + k] * y[k];
            y[i] = t / A[i * (i + 1) / 2 + i];
            mahal += y[i] * y[i];
        }
        out[(long long)B_TOTAL * AD + gr]      = -0.5f * (AD * LOG2PI + 2.f * logdiag + mahal);
        out[(long long)B_TOTAL * (AD + 1) + gr] = 0.5f * (AD * (1.f + LOG2PI) + 2.f * logdiag);
        out[(long long)B_TOTAL * (AD + 2) + gr] = valB[r];
    }
}

extern "C" void kernel_launch(void* const* d_in, const int* in_sizes, int n_in,
                              void* d_out, int out_size, void* d_ws, size_t ws_size,
                              hipStream_t stream) {
    const float* x    = (const float*)d_in[0];
    const float* act  = (const float*)d_in[1];
    const float* cW1  = (const float*)d_in[2];
    const float* cb1  = (const float*)d_in[3];
    const float* cg1  = (const float*)d_in[4];
    const float* cW2  = (const float*)d_in[5];
    const float* cb2  = (const float*)d_in[6];
    const float* cW3  = (const float*)d_in[7];
    const float* cb3  = (const float*)d_in[8];
    const float* aW1  = (const float*)d_in[9];
    const float* ab1  = (const float*)d_in[10];
    const float* aW2  = (const float*)d_in[11];
    const float* ab2  = (const float*)d_in[12];
    const float* amW  = (const float*)d_in[13];
    const float* amb  = (const float*)d_in[14];
    const float* nW1  = (const float*)d_in[15];
    const float* nb1  = (const float*)d_in[16];
    const float* ng1  = (const float*)d_in[17];
    const float* nW2  = (const float*)d_in[18];
    const float* nb2  = (const float*)d_in[19];
    const float* ng2  = (const float*)d_in[20];
    const float* nfW  = (const float*)d_in[21];
    const float* nfb  = (const float*)d_in[22];
    float* out = (float*)d_out;

    _Float16* ws = (_Float16*)d_ws;
    prep_weights<<<64, 256, 0, stream>>>(cW1, aW1, nW1, cW2, aW2, nW2, nfW, ws);
    prep_P<<<1, 256, 0, stream>>>(amW, ws + WS_P);

    fused_agent<<<B_TOTAL / 128, 128, SM_BYTES, stream>>>(
        x, act,
        ws + WS_CW1, cb1, cg1, ws + WS_CW2, cb2, cW3, cb3,
        ws + WS_AW1, ab1, ws + WS_AW2, ab2, amW, amb,
        ws + WS_NW1, nb1, ng1, ws + WS_NW2, nb2, ng2,
        ws + WS_NFW, nfb, ws + WS_P, out);
}
